// MultiheadAttention_46334107189374
// MI455X (gfx1250) — matrix operand order repaired and checked by reading, not compiled, not run on hardware
//
#include <hip/hip_runtime.h>
#include <hip/hip_bf16.h>
#include <math.h>

// Problem constants (match reference)
#define B_  4
#define L_  1024
#define E_  2048
#define H_  16
#define DH_ 128            // E/H
#define SCALE_BASE_ 512.0f

typedef __attribute__((ext_vector_type(16))) __bf16 v16bf;
typedef __attribute__((ext_vector_type(8)))  float  v8f;

// ---------------------------------------------------------------------------
// WMMA fragment loaders (contiguous k order, same for A and B, wave32)
// A 16x32 (MxK) bf16: lane m = lane&15; half = lane>>4.
//   elems 0..7  = K = half*16 + j       (VGPR0..3, 2 per VGPR)
//   elems 8..15 = K = half*16 + 8 + j   (VGPR4..7)
// B 32x16 (KxN) bf16: lane n = lane&15; elems i = K = half*16 + i.
// C/D 16x16 f32: elem i -> M = i + 8*half, N = lane&15.
// ---------------------------------------------------------------------------
__device__ inline v16bf load_a_frag(const float* __restrict__ base, int lda,
                                    int k0, int lane) {
  const int half = lane >> 4, m = lane & 15;
  const float* p = base + (size_t)m * lda + k0;
  v16bf a;
#pragma unroll
  for (int j = 0; j < 8; ++j) a[j] = (__bf16)p[half * 16 + j];
#pragma unroll
  for (int j = 0; j < 8; ++j) a[8 + j] = (__bf16)p[half * 16 + 8 + j];
  return a;
}

// B columns are contiguous rows of wrow (computes A * wrow^T)
__device__ inline v16bf load_bT_frag(const float* __restrict__ wrow, int ldw,
                                     int k0, int lane) {
  const int half = lane >> 4, n = lane & 15;
  const float* p = wrow + (size_t)n * ldw + k0 + half * 16;
  v16bf b;
#pragma unroll
  for (int i = 0; i < 16; ++i) b[i] = (__bf16)p[i];
  return b;
}

// ---------------------------------------------------------------------------
// Generic C = (A * W^T + bias) * scale.  A: MxK row-major, W: NxK row-major.
// One wave per block; each wave computes a 16(M) x 64(N) strip.
// grid = (M/16, N/64)
// ---------------------------------------------------------------------------
__global__ void gemm_xwT_kernel(const float* __restrict__ A,
                                const float* __restrict__ W,
                                const float* __restrict__ bias,
                                float* __restrict__ C,
                                int K, int N, float scale) {
  const int lane = threadIdx.x & 31;
  const int m0 = blockIdx.x * 16;
  const int n0 = blockIdx.y * 64;
  const float* arow = A + (size_t)m0 * K;

  v8f acc[4] = {v8f{}, v8f{}, v8f{}, v8f{}};
  for (int k0 = 0; k0 < K; k0 += 32) {
    __builtin_prefetch(arow + k0 + 64, 0, 1);                 // global_prefetch_b8
    v16bf a = load_a_frag(arow, K, k0, lane);
#pragma unroll
    for (int t = 0; t < 4; ++t) {
      __builtin_prefetch(W + (size_t)(n0 + t * 16) * K + k0 + 64, 0, 1);
      v16bf b = load_bT_frag(W + (size_t)(n0 + t * 16) * K, K, k0, lane);
      acc[t] = __builtin_amdgcn_wmma_f32_16x16x32_bf16(
          false, a, false, b, (short)0, acc[t], false, false);
    }
  }
  const int half = lane >> 4, nl = lane & 15;
#pragma unroll
  for (int t = 0; t < 4; ++t) {
#pragma unroll
    for (int i = 0; i < 8; ++i) {
      const int m = m0 + half * 8 + i;
      const int n = n0 + t * 16 + nl;
      C[(size_t)m * N + n] = (acc[t][i] + bias[n]) * scale;
    }
  }
}

// ---------------------------------------------------------------------------
// xPos rotary (in place on (B,L,E) with per-head pairs). One thread per pair.
// ---------------------------------------------------------------------------
__global__ void xpos_kernel(float* __restrict__ x, int downscale) {
  const size_t idx = (size_t)blockIdx.x * blockDim.x + threadIdx.x;
  const size_t total = (size_t)B_ * L_ * (E_ / 2);
  if (idx >= total) return;
  const int jcol = (int)(idx % (E_ / 2));   // pair index within E row
  const size_t row = idx / (E_ / 2);        // b*L + l
  const int l = (int)(row % L_);
  const int j = jcol % (DH_ / 2);           // pair index within head

  const float pos = (float)l - (float)(L_ / 2);
  const float base = (2.0f * (float)j + 0.4f * (float)DH_) / (1.4f * (float)DH_);
  float scale = __powf(base, pos / SCALE_BASE_);
  if (downscale) scale = 1.0f / scale;
  const float invf = __powf(10000.0f, -((float)j) / (float)(DH_ / 2));
  const float ang = (float)l * invf;
  const float sn = __sinf(ang) * scale;
  const float cs = __cosf(ang) * scale;

  float* p = x + row * E_ + 2 * (size_t)jcol;
  const float x0 = p[0], x1 = p[1];
  p[0] = x0 * cs - x1 * sn;
  p[1] = x1 * cs + x0 * sn;
}

// ---------------------------------------------------------------------------
// Scores: w[h,b,t,s] = q[bh,t,:]·k[bh,s,:] + mask[t,s] + rel[bh,t,s]
// (pre-softmax, written into the attn_weights output region, (H,B,L,L) layout)
// grid = (L/16, L/64, B*H), one wave per block, 16x64 strip, K = DH = 128.
// ---------------------------------------------------------------------------
__global__ void scores_kernel(const float* __restrict__ q,
                              const float* __restrict__ k,
                              const float* __restrict__ mask,
                              const float* __restrict__ rel,
                              float* __restrict__ wout) {
  const int lane = threadIdx.x & 31;
  const int t0 = blockIdx.x * 16;
  const int s0 = blockIdx.y * 64;
  const int bh = blockIdx.z;
  const int b = bh / H_, h = bh % H_;
  const float* qb = q + (size_t)b * L_ * E_ + (size_t)h * DH_;
  const float* kb = k + (size_t)b * L_ * E_ + (size_t)h * DH_;

  v8f acc[4] = {v8f{}, v8f{}, v8f{}, v8f{}};
#pragma unroll
  for (int k0 = 0; k0 < DH_; k0 += 32) {
    v16bf a = load_a_frag(qb + (size_t)t0 * E_, E_, k0, lane);
#pragma unroll
    for (int t = 0; t < 4; ++t) {
      v16bf bf = load_bT_frag(kb + (size_t)(s0 + t * 16) * E_, E_, k0, lane);
      acc[t] = __builtin_amdgcn_wmma_f32_16x16x32_bf16(
          false, a, false, bf, (short)0, acc[t], false, false);
    }
  }
  const size_t obase = (size_t)(h * B_ + b) * L_ * L_;
  const size_t rbase = (size_t)bh * L_ * L_;
  const int half = lane >> 4, nl = lane & 15;
#pragma unroll
  for (int t = 0; t < 4; ++t) {
#pragma unroll
    for (int i = 0; i < 8; ++i) {
      const int tt = t0 + half * 8 + i;
      const int ss = s0 + t * 16 + nl;
      const size_t off = (size_t)tt * L_ + ss;
      wout[obase + off] = acc[t][i] + mask[off] + rel[rbase + off];
    }
  }
}

// ---------------------------------------------------------------------------
// Row softmax in place. One 256-thread block per row of length L=1024.
// ---------------------------------------------------------------------------
__global__ void softmax_kernel(float* __restrict__ w) {
  float* p = w + (size_t)blockIdx.x * L_;
  const int tid = threadIdx.x;
  const int wid = tid >> 5;
  __shared__ float red[8];

  float vals[4];
  float mx = -INFINITY;
#pragma unroll
  for (int i = 0; i < 4; ++i) {
    vals[i] = p[tid + i * 256];
    mx = fmaxf(mx, vals[i]);
  }
#pragma unroll
  for (int off = 16; off; off >>= 1) mx = fmaxf(mx, __shfl_xor(mx, off, 32));
  if ((tid & 31) == 0) red[wid] = mx;
  __syncthreads();
  mx = red[0];
#pragma unroll
  for (int i = 1; i < 8; ++i) mx = fmaxf(mx, red[i]);

  float s = 0.0f;
#pragma unroll
  for (int i = 0; i < 4; ++i) {
    vals[i] = __expf(vals[i] - mx);
    s += vals[i];
  }
#pragma unroll
  for (int off = 16; off; off >>= 1) s += __shfl_xor(s, off, 32);
  __syncthreads();
  if ((tid & 31) == 0) red[wid] = s;
  __syncthreads();
  s = red[0];
#pragma unroll
  for (int i = 1; i < 8; ++i) s += red[i];
  const float inv = 1.0f / s;
#pragma unroll
  for (int i = 0; i < 4; ++i) p[tid + i * 256] = vals[i] * inv;
}

// ---------------------------------------------------------------------------
// Transpose v (B,L,E with per-head (L,DH) views) -> vT (B*H, DH, L).
// LDS 32x33 tile, both global sides coalesced.
// grid = (L/32, DH/32, B*H), block (32, 8)
// ---------------------------------------------------------------------------
__global__ void transpose_v_kernel(const float* __restrict__ v,
                                   float* __restrict__ vT) {
  __shared__ float tile[32][33];
  const int bh = blockIdx.z;
  const int b = bh / H_, h = bh % H_;
  const float* src = v + (size_t)b * L_ * E_ + (size_t)h * DH_;  // (L,DH), ld=E
  float* dst = vT + (size_t)bh * DH_ * L_;                       // (DH,L), ld=L
  const int s0 = blockIdx.x * 32;
  const int d0 = blockIdx.y * 32;
#pragma unroll
  for (int r = threadIdx.y; r < 32; r += 8)
    tile[r][threadIdx.x] = src[(size_t)(s0 + r) * E_ + d0 + threadIdx.x];
  __syncthreads();
#pragma unroll
  for (int r = threadIdx.y; r < 32; r += 8)
    dst[(size_t)(d0 + r) * L_ + s0 + threadIdx.x] = tile[threadIdx.x][r];
}

// ---------------------------------------------------------------------------
// attn[b,t,h*DH+n] = sum_s w[h,b,t,s] * vT[bh,n,s]
// grid = (L/16, DH/64, B*H); wave computes 16(t) x 64(n); K = L = 1024.
// vT rows are B columns -> contiguous b128 fragment loads.
// ---------------------------------------------------------------------------
__global__ void attnv_kernel(const float* __restrict__ w,
                             const float* __restrict__ vT,
                             float* __restrict__ attn) {
  const int lane = threadIdx.x & 31;
  const int t0 = blockIdx.x * 16;
  const int n0 = blockIdx.y * 64;
  const int bh = blockIdx.z;
  const int b = bh / H_, h = bh % H_;
  const float* wb = w + (size_t)(h * B_ + b) * L_ * L_ + (size_t)t0 * L_;
  const float* vb = vT + (size_t)bh * DH_ * L_;   // (DH, L) row-major

  v8f acc[4] = {v8f{}, v8f{}, v8f{}, v8f{}};
  for (int k0 = 0; k0 < L_; k0 += 32) {
    __builtin_prefetch(wb + k0 + 64, 0, 1);
    v16bf a = load_a_frag(wb, L_, k0, lane);
#pragma unroll
    for (int t = 0; t < 4; ++t) {
      v16bf bf = load_bT_frag(vb + (size_t)(n0 + t * 16) * L_, L_, k0, lane);
      acc[t] = __builtin_amdgcn_wmma_f32_16x16x32_bf16(
          false, a, false, bf, (short)0, acc[t], false, false);
    }
  }
  const int half = lane >> 4, nl = lane & 15;
#pragma unroll
  for (int t = 0; t < 4; ++t) {
#pragma unroll
    for (int i = 0; i < 8; ++i) {
      const int tt = t0 + half * 8 + i;
      const int col = h * DH_ + n0 + t * 16 + nl;
      attn[((size_t)b * L_ + tt) * E_ + col] = acc[t][i];
    }
  }
}

// ---------------------------------------------------------------------------
extern "C" void kernel_launch(void* const* d_in, const int* in_sizes, int n_in,
                              void* d_out, int out_size, void* d_ws,
                              size_t ws_size, hipStream_t stream) {
  (void)in_sizes; (void)n_in; (void)out_size; (void)ws_size;
  const float* query = (const float*)d_in[0];
  const float* Wq = (const float*)d_in[1];
  const float* bq = (const float*)d_in[2];
  const float* Wk = (const float*)d_in[3];
  const float* bk = (const float*)d_in[4];
  const float* Wv = (const float*)d_in[5];
  const float* bv = (const float*)d_in[6];
  const float* Wo = (const float*)d_in[7];
  const float* bo = (const float*)d_in[8];
  const float* mask = (const float*)d_in[9];
  const float* rel  = (const float*)d_in[10];

  float* out = (float*)d_out;                         // (B,L,E)
  float* out_attn = out + (size_t)B_ * L_ * E_;       // (H,B,L,L)

  const size_t BLE = (size_t)B_ * L_ * E_;
  // Workspace: 3*BLE floats (~101 MB) with buffer reuse:
  //   slot0: q  (dead after scores)  -> reused as vT
  //   slot1: k  (dead after scores)  -> reused as attn
  //   slot2: v  (dead after transpose)
  float* q    = (float*)d_ws;
  float* k    = q + BLE;
  float* v    = k + BLE;
  float* vT   = q;     // reuse
  float* attn = k;     // reuse

  const float scaling = 1.0f / sqrtf((float)DH_);
  const int M = B_ * L_;

  dim3 wave(32);
  dim3 gProj(M / 16, E_ / 64);      // (256, 32)

  // Q/K/V projections (bf16 WMMA, f32 accumulate)
  gemm_xwT_kernel<<<gProj, wave, 0, stream>>>(query, Wq, bq, q, E_, E_, scaling);
  gemm_xwT_kernel<<<gProj, wave, 0, stream>>>(query, Wk, bk, k, E_, E_, 1.0f);
  gemm_xwT_kernel<<<gProj, wave, 0, stream>>>(query, Wv, bv, v, E_, E_, 1.0f);

  // xPos rotary: q upscale, k downscale
  const size_t pairs = (size_t)B_ * L_ * (E_ / 2);
  const int xblk = 256;
  xpos_kernel<<<(unsigned)((pairs + xblk - 1) / xblk), xblk, 0, stream>>>(q, 0);
  xpos_kernel<<<(unsigned)((pairs + xblk - 1) / xblk), xblk, 0, stream>>>(k, 1);

  // scores + mask + rel_pos into attn_weights output region (pre-softmax)
  scores_kernel<<<dim3(L_ / 16, L_ / 64, B_ * H_), wave, 0, stream>>>(
      q, k, mask, rel, out_attn);

  // transpose v -> vT (q and k are dead now; vT overwrites q's slot)
  transpose_v_kernel<<<dim3(L_ / 32, DH_ / 32, B_ * H_), dim3(32, 8), 0,
                       stream>>>(v, vT);

  // row softmax in place (this IS the attn_weights output)
  softmax_kernel<<<B_ * H_ * L_, 256, 0, stream>>>(out_attn);

  // attn = w @ v (contiguous B fragments from vT)
  attnv_kernel<<<dim3(L_ / 16, DH_ / 64, B_ * H_), wave, 0, stream>>>(
      out_attn, vT, attn);

  // out = attn @ Wo^T + bo
  gemm_xwT_kernel<<<gProj, wave, 0, stream>>>(attn, Wo, bo, out, E_, E_, 1.0f);
}
